// SparseAutoencoder_28406913696395
// MI455X (gfx1250) — matrix operand order repaired and checked by reading, not compiled, not run on hardware
//
#include <hip/hip_runtime.h>
#include <hip/hip_bf16.h>

// ---------------------------------------------------------------------------
// Problem constants (from reference)
// ---------------------------------------------------------------------------
#define N_TOKENS 4096
#define D_MODEL  1024
#define D_HIDDEN 32768
#define TOPK     128
#define LN_EPS   1e-5f

// GEMM tiling
#define BM 16            // rows per block
#define BN 512           // cols per block (64 per wave, 8 waves)
#define BK 32            // K per step
#define A_PAD 8          // bf16 elems of row padding (16B) in LDS
#define B_PAD 8

typedef __bf16 bf16_t;
typedef __attribute__((ext_vector_type(16))) __bf16 v16bf;
typedef __attribute__((ext_vector_type(8)))  __bf16 v8bf;
typedef __attribute__((ext_vector_type(8)))  float  v8f;
typedef __attribute__((ext_vector_type(4)))  unsigned short us4;
typedef __attribute__((ext_vector_type(4)))  int v4i;

// ---------------------------------------------------------------------------
// CDNA5 async global->LDS copy (ASYNCcnt path), with compile-safe fallback
// ---------------------------------------------------------------------------
#if defined(__has_builtin)
#if __has_builtin(__builtin_amdgcn_global_load_async_to_lds_b128)
#define HAVE_ASYNC_LDS 1
#endif
#endif

__device__ __forceinline__ void copy16_g2lds(const unsigned short* __restrict__ g,
                                             unsigned short* l) {
#ifdef HAVE_ASYNC_LDS
    __builtin_amdgcn_global_load_async_to_lds_b128(
        (__attribute__((address_space(1))) v4i*)g,
        (__attribute__((address_space(3))) v4i*)l, 0, 0);
#else
    *(uint4*)l = *(const uint4*)g;
#endif
}

__device__ __forceinline__ void wait_async_zero() {
#ifdef HAVE_ASYNC_LDS
#if __has_builtin(__builtin_amdgcn_s_wait_asynccnt)
    __builtin_amdgcn_s_wait_asynccnt(0);
#else
    asm volatile("s_wait_asynccnt 0x0" ::: "memory");
#endif
#endif
}

// fp32 -> bf16 round-to-nearest-even (bit trick; storage via ushort)
__device__ __forceinline__ unsigned short f32_to_bf16_rne(float f) {
    unsigned u = __float_as_uint(f);
    unsigned r = u + 0x7FFFu + ((u >> 16) & 1u);
    return (unsigned short)(r >> 16);
}

// ---------------------------------------------------------------------------
// Kernel 1: LayerNorm (torch-style: unbiased std, divide by (std+eps)),
//           subtract b_pre, emit bf16 xn + per-row mu/std.
// ---------------------------------------------------------------------------
__global__ void __launch_bounds__(256) ln_kernel(
    const float* __restrict__ x, const float* __restrict__ b_pre,
    unsigned short* __restrict__ xn, float* __restrict__ mu_out,
    float* __restrict__ std_out)
{
    const int row = blockIdx.x;
    const int tid = threadIdx.x;
    const float4 v = *(const float4*)(x + (size_t)row * D_MODEL + tid * 4);

    __shared__ float s1[256], s2[256];
    s1[tid] = v.x + v.y + v.z + v.w;
    s2[tid] = v.x * v.x + v.y * v.y + v.z * v.z + v.w * v.w;
    __syncthreads();
    for (int off = 128; off > 0; off >>= 1) {
        if (tid < off) { s1[tid] += s1[tid + off]; s2[tid] += s2[tid + off]; }
        __syncthreads();
    }
    __shared__ float smu, sinv;
    if (tid == 0) {
        float mu  = s1[0] * (1.0f / D_MODEL);
        float var = (s2[0] - (float)D_MODEL * mu * mu) * (1.0f / (D_MODEL - 1));
        float sd  = sqrtf(fmaxf(var, 0.0f));
        smu = mu; sinv = 1.0f / (sd + LN_EPS);
        mu_out[row]  = mu;
        std_out[row] = sd;
    }
    __syncthreads();

    const float4 bp = *(const float4*)(b_pre + tid * 4);
    us4 o;
    o.x = f32_to_bf16_rne((v.x - smu) * sinv - bp.x);
    o.y = f32_to_bf16_rne((v.y - smu) * sinv - bp.y);
    o.z = f32_to_bf16_rne((v.z - smu) * sinv - bp.z);
    o.w = f32_to_bf16_rne((v.w - smu) * sinv - bp.w);
    *(us4*)(xn + (size_t)row * D_MODEL + tid * 4) = o;
}

// ---------------------------------------------------------------------------
// Kernel 2: w_enc fp32 -> bf16 (one-time; halves L2 footprint of the GEMM)
// ---------------------------------------------------------------------------
__global__ void __launch_bounds__(256) cvt_bf16_kernel(
    const float* __restrict__ in, unsigned short* __restrict__ out)
{
    const size_t i = ((size_t)blockIdx.x * 256 + threadIdx.x) * 4;
    const float4 v = *(const float4*)(in + i);
    us4 o;
    o.x = f32_to_bf16_rne(v.x);
    o.y = f32_to_bf16_rne(v.y);
    o.z = f32_to_bf16_rne(v.z);
    o.w = f32_to_bf16_rne(v.w);
    *(us4*)(out + i) = o;
}

// ---------------------------------------------------------------------------
// Kernel 3: encode GEMM via V_WMMA_F32_16X16X32_BF16 with async LDS staging.
//
// Block = 256 threads (8 waves). Block tile: 16(M) x 512(N); each wave owns a
// 16x64 slice (4 f32 accumulators). A tile (16x1024 bf16) staged to LDS once;
// B tile (32x512 bf16) double-buffered via GLOBAL_LOAD_ASYNC_TO_LDS_B128 so
// the next K-step's transfer overlaps the current step's WMMAs; sync per step
// with s_wait_asynccnt 0 + workgroup barrier.
//
// WMMA VGPR layouts per ISA 7.12.2:
//  A (16x32 bf16): lanes 0-15 -> M=lane, elems {K..K+7, K+16..K+23};
//                  lanes 16-31 -> M=lane-16, elems {K+8..K+15, K+24..K+31}.
//  B (32x16 bf16): N=lane&15, elems -> K in the same order as A's lane half.
//  C (16x16 f32):  VGPR r -> M=r (lanes 0-15) / M=r+8 (lanes 16-31), N=lane&15.
// ---------------------------------------------------------------------------
__global__ void __launch_bounds__(256) encode_gemm_kernel(
    const unsigned short* __restrict__ xn,
    const unsigned short* __restrict__ wenc,
    const float* __restrict__ b_enc,
    float* __restrict__ pre)
{
    __shared__ unsigned short sA[BM][D_MODEL + A_PAD];     // ~33 KB
    __shared__ unsigned short sB[2][BK][BN + B_PAD];       // ~65 KB (double buf)

    const int tid  = threadIdx.x;
    const int lane = tid & 31;
    const int wave = tid >> 5;
    const int m0   = blockIdx.x * BM;
    const int nb   = blockIdx.y * BN;
    const int l    = lane & 15;
    const bool hiH = lane >= 16;

    // ---- stage A tile once: 16 x 1024 bf16 = 2048 chunks of 16B ----
#pragma unroll
    for (int i = 0; i < 8; ++i) {
        const int c   = tid + i * 256;
        const int row = c >> 7;          // 128 x 16B chunks per row
        const int cc  = c & 127;
        copy16_g2lds(xn + (size_t)(m0 + row) * D_MODEL + cc * 8,
                     &sA[row][cc * 8]);
    }
    // ---- stage B tile for kb=0 into buffer 0: 32 x 512 bf16 = 2048 chunks --
#pragma unroll
    for (int i = 0; i < 8; ++i) {
        const int c   = tid + i * 256;
        const int row = c >> 6;          // 64 x 16B chunks per row
        const int cc  = c & 63;
        copy16_g2lds(wenc + (size_t)row * D_HIDDEN + nb + cc * 8,
                     &sB[0][row][cc * 8]);
    }
    wait_async_zero();
    __syncthreads();

    v8f c0 = {}, c1 = {}, c2 = {}, c3 = {};
    const int aoff = hiH ? 8 : 0;

#pragma unroll 2
    for (int kb = 0; kb < D_MODEL / BK; ++kb) {
        const int buf = kb & 1;
        const int k0  = kb * BK;

        // issue async B loads for the next K-step into the other buffer
        if (kb + 1 < D_MODEL / BK) {
#pragma unroll
            for (int i = 0; i < 8; ++i) {
                const int c   = tid + i * 256;
                const int row = c >> 6;
                const int cc  = c & 63;
                copy16_g2lds(wenc + (size_t)(k0 + BK + row) * D_HIDDEN + nb + cc * 8,
                             &sB[buf ^ 1][row][cc * 8]);
            }
        }

        // A fragment from LDS (two 16B chunks per lane)
        v8bf alo = *(const v8bf*)&sA[l][k0 + aoff];
        v8bf ahi = *(const v8bf*)&sA[l][k0 + aoff + 16];
        v16bf a = __builtin_shufflevector(alo, ahi,
                    0, 1, 2, 3, 4, 5, 6, 7, 8, 9, 10, 11, 12, 13, 14, 15);

        // B fragments from LDS: lane -> N column (l), elems -> K in A's order, 4 groups of 16 columns
        const int bcol = wave * 64 + l;
        v16bf b0, b1, b2, b3;
#pragma unroll
        for (int s = 0; s < 16; ++s) {
            const int kr = aoff + (s & 7) + ((s & 8) ? 16 : 0);
            const unsigned short* br = &sB[buf][kr][bcol];
            b0[s] = __builtin_bit_cast(__bf16, br[0]);
            b1[s] = __builtin_bit_cast(__bf16, br[16]);
            b2[s] = __builtin_bit_cast(__bf16, br[32]);
            b3[s] = __builtin_bit_cast(__bf16, br[48]);
        }

        c0 = __builtin_amdgcn_wmma_f32_16x16x32_bf16(false, a, false, b0, (short)0, c0, false, false);
        c1 = __builtin_amdgcn_wmma_f32_16x16x32_bf16(false, a, false, b1, (short)0, c1, false, false);
        c2 = __builtin_amdgcn_wmma_f32_16x16x32_bf16(false, a, false, b2, (short)0, c2, false, false);
        c3 = __builtin_amdgcn_wmma_f32_16x16x32_bf16(false, a, false, b3, (short)0, c3, false, false);

        // next buffer must be fully written (my loads done + everyone arrived)
        wait_async_zero();
        __syncthreads();
    }

    // Epilogue: + b_enc, scatter C per VGPR layout
    const int rbase = m0 + (hiH ? 8 : 0);
    const int n0 = nb + wave * 64;
    v8f acc[4] = { c0, c1, c2, c3 };
#pragma unroll
    for (int j = 0; j < 4; ++j) {
        const int col = n0 + j * 16 + l;
        const float be = b_enc[col];
#pragma unroll
        for (int r = 0; r < 8; ++r) {
            pre[(size_t)(rbase + r) * D_HIDDEN + col] = acc[j][r] + be;
        }
    }
}

// ---------------------------------------------------------------------------
// Kernel 4: per-row top-K (K=128 of 32768) + ReLU via 2048-bin radix histogram
// on the monotone fp32 key, suffix-scan threshold, one emit re-scan.
// ---------------------------------------------------------------------------
__global__ void __launch_bounds__(256) topk_kernel(
    const float* __restrict__ pre,
    float* __restrict__ vals, int* __restrict__ idxs)
{
    const int row = blockIdx.x;
    const int tid = threadIdx.x;
    const float* __restrict__ p = pre + (size_t)row * D_HIDDEN;

    __shared__ unsigned hist[2048];
    __shared__ unsigned chunk[256];
    __shared__ int sT, sAbove, sNeed;
    __shared__ unsigned cHi, cTie;

    for (int i = tid; i < 2048; i += 256) hist[i] = 0;
    if (tid == 0) { cHi = 0; cTie = 0; }
    __syncthreads();

    for (int i = tid; i < D_HIDDEN; i += 256) {
        unsigned u = __float_as_uint(p[i]);
        unsigned key = (u & 0x80000000u) ? ~u : (u | 0x80000000u);
        atomicAdd(&hist[key >> 21], 1u);
    }
    __syncthreads();

    unsigned cs = 0;
#pragma unroll
    for (int j = 0; j < 8; ++j) cs += hist[tid * 8 + j];
    chunk[tid] = cs;
    __syncthreads();

    if (tid == 0) {
        unsigned acc = 0;
        int c = 255;
        for (; c > 0; --c) {
            if (acc + chunk[c] >= (unsigned)TOPK) break;
            acc += chunk[c];
        }
        unsigned above = acc;
        int T = c * 8;
        for (int b = c * 8 + 7; b >= c * 8; --b) {
            if (above + hist[b] >= (unsigned)TOPK || b == c * 8) { T = b; break; }
            above += hist[b];
        }
        sT = T; sAbove = (int)above; sNeed = TOPK - (int)above;
    }
    __syncthreads();

    const int T = sT, above = sAbove, need = sNeed;
    for (int i = tid; i < D_HIDDEN; i += 256) {
        const float v = p[i];
        unsigned u = __float_as_uint(v);
        unsigned key = (u & 0x80000000u) ? ~u : (u | 0x80000000u);
        const int b = (int)(key >> 21);
        if (b > T) {
            unsigned s = atomicAdd(&cHi, 1u);
            vals[(size_t)row * TOPK + s] = fmaxf(v, 0.0f);   // ReLU
            idxs[(size_t)row * TOPK + s] = i;
        } else if (b == T) {
            unsigned t = atomicAdd(&cTie, 1u);
            if ((int)t < need) {
                unsigned s = (unsigned)above + t;
                vals[(size_t)row * TOPK + s] = fmaxf(v, 0.0f);
                idxs[(size_t)row * TOPK + s] = i;
            }
        }
    }
}

// ---------------------------------------------------------------------------
// Kernel 5: sparse decode + un-LayerNorm.
// out = (sum_k val_k * w_dec[idx_k, :] + b_pre) * std + mu
// ---------------------------------------------------------------------------
__global__ void __launch_bounds__(256) decode_kernel(
    const float* __restrict__ vals, const int* __restrict__ idxs,
    const float* __restrict__ wdec, const float* __restrict__ b_pre,
    const float* __restrict__ mu, const float* __restrict__ stdv,
    float* __restrict__ out)
{
    const int row = blockIdx.x;
    __shared__ float sv[TOPK];
    __shared__ int   si[TOPK];
    if (threadIdx.x < TOPK) {
        sv[threadIdx.x] = vals[(size_t)row * TOPK + threadIdx.x];
        si[threadIdx.x] = idxs[(size_t)row * TOPK + threadIdx.x];
    }
    __syncthreads();

    const int c0 = threadIdx.x * 4;
    float ax = 0.f, ay = 0.f, az = 0.f, aw = 0.f;
#pragma unroll 4
    for (int j = 0; j < TOPK; ++j) {
        if (j + 8 < TOPK)  // warm L2 for the gather 8 rows ahead
            __builtin_prefetch((const void*)(wdec + (size_t)si[j + 8] * D_MODEL + c0), 0, 1);
        const float v = sv[j];
        const float4 w = *(const float4*)(wdec + (size_t)si[j] * D_MODEL + c0);
        ax = fmaf(v, w.x, ax);
        ay = fmaf(v, w.y, ay);
        az = fmaf(v, w.z, az);
        aw = fmaf(v, w.w, aw);
    }
    const float m = mu[row], sd = stdv[row];
    const float4 bp = *(const float4*)(b_pre + c0);
    float4 o;
    o.x = fmaf(ax + bp.x, sd, m);
    o.y = fmaf(ay + bp.y, sd, m);
    o.z = fmaf(az + bp.z, sd, m);
    o.w = fmaf(aw + bp.w, sd, m);
    *(float4*)(out + (size_t)row * D_MODEL + c0) = o;
}

// ---------------------------------------------------------------------------
// Host-side launch
// ---------------------------------------------------------------------------
extern "C" void kernel_launch(void* const* d_in, const int* in_sizes, int n_in,
                              void* d_out, int out_size, void* d_ws, size_t ws_size,
                              hipStream_t stream) {
    (void)in_sizes; (void)n_in; (void)out_size; (void)ws_size;

    const float* x     = (const float*)d_in[0];   // [4096,1024]
    const float* w_enc = (const float*)d_in[1];   // [1024,32768]
    const float* w_dec = (const float*)d_in[2];   // [32768,1024]
    const float* b_enc = (const float*)d_in[3];   // [32768]
    const float* b_pre = (const float*)d_in[4];   // [1024]
    float* out = (float*)d_out;                   // [4096,1024]

    // workspace layout
    char* ws = (char*)d_ws;
    const size_t XN_OFF   = 0;                                                  // 8 MB bf16 xn
    const size_t MU_OFF   = XN_OFF  + (size_t)N_TOKENS * D_MODEL * 2;           // 16 KB
    const size_t STD_OFF  = MU_OFF  + (size_t)N_TOKENS * 4;                     // 16 KB
    const size_t WENC_OFF = STD_OFF + (size_t)N_TOKENS * 4;                     // 64 MB bf16 w_enc
    const size_t PRE_OFF  = WENC_OFF + (size_t)D_MODEL * D_HIDDEN * 2;          // 512 MB f32 pre_acts
    const size_t VAL_OFF  = PRE_OFF + (size_t)N_TOKENS * D_HIDDEN * 4;          // 2 MB
    const size_t IDX_OFF  = VAL_OFF + (size_t)N_TOKENS * TOPK * 4;              // 2 MB

    unsigned short* xn_bf   = (unsigned short*)(ws + XN_OFF);
    float*          mu_buf  = (float*)(ws + MU_OFF);
    float*          std_buf = (float*)(ws + STD_OFF);
    unsigned short* wenc_bf = (unsigned short*)(ws + WENC_OFF);
    float*          pre     = (float*)(ws + PRE_OFF);
    float*          vals    = (float*)(ws + VAL_OFF);
    int*            idxs    = (int*)(ws + IDX_OFF);

    ln_kernel<<<N_TOKENS, 256, 0, stream>>>(x, b_pre, xn_bf, mu_buf, std_buf);

    cvt_bf16_kernel<<<(D_MODEL * (size_t)D_HIDDEN) / (256 * 4), 256, 0, stream>>>(w_enc, wenc_bf);

    dim3 ggrid(N_TOKENS / BM, D_HIDDEN / BN);
    encode_gemm_kernel<<<ggrid, 256, 0, stream>>>(xn_bf, wenc_bf, b_enc, pre);

    topk_kernel<<<N_TOKENS, 256, 0, stream>>>(pre, vals, idxs);

    decode_kernel<<<N_TOKENS, 256, 0, stream>>>(vals, idxs, w_dec, b_pre,
                                                mu_buf, std_buf, out);
}
